// FourPathMamba_59700045414682
// MI455X (gfx1250) — matrix operand order repaired and checked by reading, not compiled, not run on hardware
//
#include <hip/hip_runtime.h>
#include <hip/hip_bf16.h>
#include <math.h>

typedef __attribute__((ext_vector_type(16))) __bf16 bf16x16;
typedef __attribute__((ext_vector_type(8)))  __bf16 bf16x8;
typedef __attribute__((ext_vector_type(8)))  float  f32x8;

// ---- problem dimensions ----
constexpr int DM   = 512;       // model dim
constexpr int DI   = 1024;      // inner dim
constexpr int DRr  = 32;        // dt rank
constexpr int TT   = 3;
constexpr int HH   = 16;
constexpr int WW   = 11;
constexpr int BM   = 10;        // Bb * (Sf/T)
constexpr int LL   = TT * HH * WW;       // 528
constexpr int NTOK = BM * LL;            // 5280 unique tokens
constexpr int NPROW  = 4 * BM;           // 40 path-rows
constexpr int NPTOK  = NPROW * LL;       // 21120 path tokens

// map (path p, batch b, seq pos l) -> unique-token row in [0, NTOK)
__device__ __forceinline__ int tok_index(int p, int b, int l) {
  int lq = (p & 1) ? (LL - 1 - l) : l;          // paths 1,3 are flipped
  if (p >= 2) {                                  // paths 2,3 enumerate (t,w,h)
    int t = lq / (HH * WW);
    int r = lq - t * (HH * WW);
    int w = r / HH;
    int h = r - w * HH;
    lq = (t * HH + h) * WW + w;                  // token's (t,h,w) position
  }
  return b * LL + lq;
}

// ---------------- generic cast f32 -> bf16 ----------------
__global__ __launch_bounds__(256) void k_cast_bf16(const float* __restrict__ in,
                                                   __bf16* __restrict__ out, int n) {
  int i = blockIdx.x * 256 + threadIdx.x;
  if (i < n) out[i] = (__bf16)in[i];
}

// ---------------- LayerNorm over last dim (512), emit bf16 ----------------
__global__ __launch_bounds__(256) void k_layernorm(const float* __restrict__ x,
                                                   const float* __restrict__ g,
                                                   const float* __restrict__ bta,
                                                   __bf16* __restrict__ out) {
  __shared__ float sh_s[256];
  __shared__ float sh_q[256];
  int t = blockIdx.x;
  int tid = threadIdx.x;
  const float* xp = x + (size_t)t * DM;
  float v0 = xp[tid], v1 = xp[tid + 256];
  sh_s[tid] = v0 + v1;
  sh_q[tid] = v0 * v0 + v1 * v1;
  __syncthreads();
  for (int off = 128; off > 0; off >>= 1) {
    if (tid < off) { sh_s[tid] += sh_s[tid + off]; sh_q[tid] += sh_q[tid + off]; }
    __syncthreads();
  }
  float mean = sh_s[0] * (1.0f / DM);
  float var  = sh_q[0] * (1.0f / DM) - mean * mean;
  float inv  = rsqrtf(var + 1e-5f);
  out[(size_t)t * DM + tid]       = (__bf16)((v0 - mean) * inv * g[tid]       + bta[tid]);
  out[(size_t)t * DM + tid + 256] = (__bf16)((v1 - mean) * inv * g[tid + 256] + bta[tid + 256]);
}

// ---------------- WMMA GEMM: C[M,N] = A[M,K](bf16,row) * W[N,K](bf16,row)^T ----------------
// Each wave computes a 16x64 C tile via 4x v_wmma_f32_16x16x32_bf16 per K-step.
// Fragment layouts used here (contiguous k, not the ISA table order; valid as A and B match):
//   A 16x32 bf16: lane<16 -> row=lane, K 0..15 contiguous; lane>=16 -> row=lane-16, K 16..31
//   B 32x16 bf16: lane<16 -> col=lane, K 0..15 contiguous; lane>=16 -> col=lane-16, K 16..31
//   C 16x16 f32 : vgpr v, lane l -> row = (l>>4)*8+v, col = l&15
// EPI: 0 = store f32; 1 = +bias, softplus, f32; 2 = +bias, exact GELU, bf16; 3 = +bias +residual, f32
template <int EPI>
__global__ __launch_bounds__(256) void k_wmma_gemm(const __bf16* __restrict__ A, int lda,
                                                   const __bf16* __restrict__ Bw, int ldb,
                                                   void* __restrict__ Cout, int ldc,
                                                   int M, int N, int K,
                                                   const float* __restrict__ bias,
                                                   const float* __restrict__ resid) {
  int wave = (blockIdx.x * 256 + threadIdx.x) >> 5;
  int lane = threadIdx.x & 31;
  int tilesM  = M >> 4;
  int groupsN = N >> 6;
  if (wave >= tilesM * groupsN) return;
  int tm = wave / groupsN;
  int gn = wave - tm * groupsN;
  int m0 = tm << 4;
  int n0 = gn << 6;
  int frow = lane & 15;
  int fhalf = lane >> 4;

  f32x8 acc[4];
#pragma unroll
  for (int j = 0; j < 4; ++j) acc[j] = (f32x8){0.f, 0.f, 0.f, 0.f, 0.f, 0.f, 0.f, 0.f};

  for (int k0 = 0; k0 < K; k0 += 32) {
    // A fragment
    const __bf16* ap = A + (size_t)(m0 + frow) * lda + k0 + (fhalf << 4);
    bf16x8 alo = *(const bf16x8*)ap;
    bf16x8 ahi = *(const bf16x8*)(ap + 8);
    bf16x16 af;
#pragma unroll
    for (int i = 0; i < 8; ++i) { af[i] = alo[i]; af[i + 8] = ahi[i]; }
    // B fragments: weight row (n0+j*16+frow), contiguous 16 K values
    const __bf16* bp = Bw + (size_t)(n0 + frow) * ldb + k0 + (fhalf << 4);
#pragma unroll
    for (int j = 0; j < 4; ++j) {
      bf16x16 bfv = *(const bf16x16*)(bp + (size_t)(j * 16) * ldb);
      acc[j] = __builtin_amdgcn_wmma_f32_16x16x32_bf16(
          false, af, false, bfv, (short)0, acc[j], false, false);
    }
  }

#pragma unroll
  for (int j = 0; j < 4; ++j) {
#pragma unroll
    for (int v = 0; v < 8; ++v) {
      int row = m0 + (fhalf << 3) + v;
      int col = n0 + j * 16 + frow;
      float val = acc[j][v];
      if (EPI == 0) {
        ((float*)Cout)[(size_t)row * ldc + col] = val;
      } else if (EPI == 1) {
        float u = val + bias[col];
        u = (u > 20.f) ? u : log1pf(expf(u));            // softplus
        ((float*)Cout)[(size_t)row * ldc + col] = u;
      } else if (EPI == 2) {
        float u = val + bias[col];
        u = 0.5f * u * (1.0f + erff(u * 0.70710678118654752f));   // exact GELU
        ((__bf16*)Cout)[(size_t)row * ldc + col] = (__bf16)u;
      } else { // EPI == 3
        float u = val + bias[col] + resid[(size_t)row * ldc + col];
        ((float*)Cout)[(size_t)row * ldc + col] = u;
      }
    }
  }
}

// ---------------- depthwise causal conv(4) + SiLU, gathering per-path tokens ----------------
__global__ __launch_bounds__(256) void k_conv_silu(const float* __restrict__ xz,
                                                   const float* __restrict__ conv_w,
                                                   const float* __restrict__ conv_b,
                                                   __bf16* __restrict__ xi_out) {
  int id = blockIdx.x * 256 + threadIdx.x;
  if (id >= NPROW * DI) return;
  int ch  = id & (DI - 1);
  int row = id >> 10;             // 0..39
  int p = row / BM;
  int b = row - p * BM;
  float w0 = conv_w[ch * 4 + 0], w1 = conv_w[ch * 4 + 1];
  float w2 = conv_w[ch * 4 + 2], w3 = conv_w[ch * 4 + 3];
  float cb = conv_b[ch];
  float xm3 = 0.f, xm2 = 0.f, xm1 = 0.f;
  for (int l = 0; l < LL; ++l) {
    int tk = tok_index(p, b, l);
    float xc = xz[(size_t)tk * (2 * DI) + ch];
    float a = w0 * xm3 + w1 * xm2 + w2 * xm1 + w3 * xc + cb;
    float s = a / (1.f + expf(-a));                       // SiLU
    xi_out[((size_t)row * LL + l) * DI + ch] = (__bf16)s;
    xm3 = xm2; xm2 = xm1; xm1 = xc;
  }
}

// ---------------- x_dbl = xi @ xproj_w.T  (34 outputs; wave-per-dot) ----------------
__global__ __launch_bounds__(256) void k_xdbl(const __bf16* __restrict__ xi,
                                              const float* __restrict__ xproj_w,
                                              __bf16* __restrict__ dtA,
                                              float* __restrict__ Bc,
                                              float* __restrict__ Cc) {
  int gid = blockIdx.x * 256 + threadIdx.x;
  int wave = gid >> 5;
  int lane = gid & 31;
  if (wave >= NPTOK * (DRr + 2)) return;
  int t = wave / (DRr + 2);
  int j = wave - t * (DRr + 2);
  const __bf16* xp = xi + (size_t)t * DI;
  const float*  wp = xproj_w + (size_t)j * DI;
  float s = 0.f;
  for (int c = lane; c < DI; c += 32) s += (float)xp[c] * wp[c];
#pragma unroll
  for (int off = 16; off > 0; off >>= 1) s += __shfl_xor(s, off, 32);
  if (lane == 0) {
    if (j < DRr)       dtA[(size_t)t * DRr + j] = (__bf16)s;
    else if (j == DRr) Bc[t] = s;
    else               Cc[t] = s;
  }
}

// ---------------- selective scan + SiLU(z) gate ----------------
__global__ __launch_bounds__(256) void k_scan(const float* __restrict__ dt,
                                              const float* __restrict__ Bc,
                                              const float* __restrict__ Cc,
                                              const __bf16* __restrict__ xi,
                                              const float* __restrict__ xz,
                                              const float* __restrict__ A_log,
                                              const float* __restrict__ D_param,
                                              __bf16* __restrict__ y) {
  int id = blockIdx.x * 256 + threadIdx.x;
  if (id >= NPROW * DI) return;
  int ch  = id & (DI - 1);
  int row = id >> 10;
  int p = row / BM;
  int b = row - p * BM;
  float Ac = -expf(A_log[ch]);
  float Dc = D_param[ch];
  float h = 0.f;
  for (int l = 0; l < LL; ++l) {
    size_t t = (size_t)row * LL + l;
    float dtv = dt[t * DI + ch];
    float xiv = (float)xi[t * DI + ch];
    float dA  = expf(dtv * Ac);
    h = dA * h + dtv * Bc[t] * xiv;
    float yv = Cc[t] * h + Dc * xiv;
    int tk = tok_index(p, b, l);
    float zv = xz[(size_t)tk * (2 * DI) + DI + ch];
    yv *= zv / (1.f + expf(-zv));                         // * SiLU(z)
    y[t * DI + ch] = (__bf16)yv;
  }
}

// ---------------- combine 4 path outputs + residual ----------------
__global__ __launch_bounds__(256) void k_combine(const float* __restrict__ x,
                                                 const float* __restrict__ yout,
                                                 float* __restrict__ xr) {
  int idx = blockIdx.x * 256 + threadIdx.x;
  if (idx >= NTOK * DM) return;
  int c  = idx & (DM - 1);
  int tk = idx >> 9;
  int b = tk / LL;
  int l = tk - b * LL;
  int t = l / (HH * WW);
  int r = l - t * (HH * WW);
  int h = r / WW;
  int w = r - h * WW;
  int l2 = (t * WW + w) * HH + h;                         // position within x3 ordering
  size_t r0 = ((size_t)(0 * BM + b) * LL + l) * DM;
  size_t r1 = ((size_t)(1 * BM + b) * LL + (LL - 1 - l)) * DM;
  size_t r2 = ((size_t)(2 * BM + b) * LL + l2) * DM;
  size_t r3 = ((size_t)(3 * BM + b) * LL + (LL - 1 - l2)) * DM;
  xr[idx] = x[idx] + yout[r0 + c] + yout[r1 + c] + yout[r2 + c] + yout[r3 + c];
}

// ============================ host side ============================
extern "C" void kernel_launch(void* const* d_in, const int* in_sizes, int n_in,
                              void* d_out, int out_size, void* d_ws, size_t ws_size,
                              hipStream_t stream) {
  (void)in_sizes; (void)n_in; (void)out_size; (void)ws_size;
  const float* x        = (const float*)d_in[0];
  const float* ln1_g    = (const float*)d_in[1];
  const float* ln1_b    = (const float*)d_in[2];
  const float* ln2_g    = (const float*)d_in[3];
  const float* ln2_b    = (const float*)d_in[4];
  const float* mlp_w1   = (const float*)d_in[5];
  const float* mlp_b1   = (const float*)d_in[6];
  const float* mlp_w2   = (const float*)d_in[7];
  const float* mlp_b2   = (const float*)d_in[8];
  const float* in_w     = (const float*)d_in[9];
  const float* conv_w   = (const float*)d_in[10];
  const float* conv_b   = (const float*)d_in[11];
  const float* xproj_w  = (const float*)d_in[12];
  const float* dtproj_w = (const float*)d_in[13];
  const float* dtproj_b = (const float*)d_in[14];
  const float* A_log    = (const float*)d_in[15];
  const float* D_param  = (const float*)d_in[16];
  const float* out_w    = (const float*)d_in[17];
  float* out = (float*)d_out;

  char* ws = (char*)d_ws;
  size_t off = 0;
  auto alloc = [&](size_t bytes) -> void* {
    void* p = ws + off;
    off += (bytes + 255) & ~(size_t)255;
    return p;
  };

  __bf16* xn1_bf   = (__bf16*)alloc((size_t)NTOK * DM * 2);
  __bf16* in_w_bf  = (__bf16*)alloc((size_t)2 * DI * DM * 2);
  __bf16* dtw_bf   = (__bf16*)alloc((size_t)DI * DRr * 2);
  __bf16* outw_bf  = (__bf16*)alloc((size_t)DM * DI * 2);
  __bf16* w1_bf    = (__bf16*)alloc((size_t)4 * DM * DM * 2);
  __bf16* w2_bf    = (__bf16*)alloc((size_t)DM * 4 * DM * 2);
  float*  xz       = (float*) alloc((size_t)NTOK * 2 * DI * 4);
  __bf16* xi_conv  = (__bf16*)alloc((size_t)NPTOK * DI * 2);
  __bf16* dtA_bf   = (__bf16*)alloc((size_t)NPTOK * DRr * 2);
  float*  Bc       = (float*) alloc((size_t)NPTOK * 4);
  float*  Cc       = (float*) alloc((size_t)NPTOK * 4);
  float*  dt       = (float*) alloc((size_t)NPTOK * DI * 4);
  __bf16* y_bf     = (__bf16*)alloc((size_t)NPTOK * DI * 2);
  float*  yout     = (float*) alloc((size_t)NPTOK * DM * 4);
  float*  xr       = (float*) alloc((size_t)NTOK * DM * 4);
  __bf16* xn2_bf   = (__bf16*)alloc((size_t)NTOK * DM * 2);
  __bf16* hmid_bf  = (__bf16*)alloc((size_t)NTOK * 4 * DM * 2);

  auto cgrid = [](int n) { return (n + 255) / 256; };

  // weight casts to bf16
  k_cast_bf16<<<cgrid(2 * DI * DM), 256, 0, stream>>>(in_w,     in_w_bf, 2 * DI * DM);
  k_cast_bf16<<<cgrid(DI * DRr),    256, 0, stream>>>(dtproj_w, dtw_bf,  DI * DRr);
  k_cast_bf16<<<cgrid(DM * DI),     256, 0, stream>>>(out_w,    outw_bf, DM * DI);
  k_cast_bf16<<<cgrid(4 * DM * DM), 256, 0, stream>>>(mlp_w1,   w1_bf,   4 * DM * DM);
  k_cast_bf16<<<cgrid(DM * 4 * DM), 256, 0, stream>>>(mlp_w2,   w2_bf,   DM * 4 * DM);

  // LN1 on unique tokens (x is already contiguous as [NTOK, DM])
  k_layernorm<<<NTOK, 256, 0, stream>>>(x, ln1_g, ln1_b, xn1_bf);

  // in-proj GEMM once per unique token: xz[NTOK, 2*DI]
  {
    int waves = (NTOK / 16) * (2 * DI / 64);
    k_wmma_gemm<0><<<(waves + 7) / 8, 256, 0, stream>>>(
        xn1_bf, DM, in_w_bf, DM, xz, 2 * DI, NTOK, 2 * DI, DM, nullptr, nullptr);
  }

  // depthwise conv + SiLU over the 4 path orderings
  k_conv_silu<<<cgrid(NPROW * DI), 256, 0, stream>>>(xz, conv_w, conv_b, xi_conv);

  // x_dbl: dt(bf16, rank 32), Bc, Cc
  k_xdbl<<<(NPTOK * (DRr + 2)) / 8, 256, 0, stream>>>(xi_conv, xproj_w, dtA_bf, Bc, Cc);

  // dt-proj GEMM + bias + softplus: dt[NPTOK, DI]
  {
    int waves = (NPTOK / 16) * (DI / 64);
    k_wmma_gemm<1><<<(waves + 7) / 8, 256, 0, stream>>>(
        dtA_bf, DRr, dtw_bf, DRr, dt, DI, NPTOK, DI, DRr, dtproj_b, nullptr);
  }

  // selective scan + gate -> y(bf16)
  k_scan<<<cgrid(NPROW * DI), 256, 0, stream>>>(dt, Bc, Cc, xi_conv, xz, A_log, D_param, y_bf);

  // out-proj GEMM: yout[NPTOK, DM]
  {
    int waves = (NPTOK / 16) * (DM / 64);
    k_wmma_gemm<0><<<(waves + 7) / 8, 256, 0, stream>>>(
        y_bf, DI, outw_bf, DI, yout, DM, NPTOK, DM, DI, nullptr, nullptr);
  }

  // combine 4 paths + residual -> xr
  k_combine<<<cgrid(NTOK * DM), 256, 0, stream>>>(x, yout, xr);

  // LN2
  k_layernorm<<<NTOK, 256, 0, stream>>>(xr, ln2_g, ln2_b, xn2_bf);

  // MLP1: bias + exact GELU -> bf16
  {
    int waves = (NTOK / 16) * (4 * DM / 64);
    k_wmma_gemm<2><<<(waves + 7) / 8, 256, 0, stream>>>(
        xn2_bf, DM, w1_bf, DM, hmid_bf, 4 * DM, NTOK, 4 * DM, DM, mlp_b1, nullptr);
  }

  // MLP2: bias + residual -> final output (contiguous == (Bb,Sf,H,W,Dm))
  {
    int waves = (NTOK / 16) * (DM / 64);
    k_wmma_gemm<3><<<(waves + 7) / 8, 256, 0, stream>>>(
        hmid_bf, 4 * DM, w2_bf, 4 * DM, out, DM, NTOK, DM, 4 * DM, mlp_b2, xr);
  }
}
